// Model_78408922956275
// MI455X (gfx1250) — matrix operand order repaired and checked by reading, not compiled, not run on hardware
//
#include <hip/hip_runtime.h>
#include <math.h>

// ---------------- model constants ----------------
constexpr int kB  = 8;
constexpr int kS  = 256;
constexpr int kH  = 512;
constexpr int kNH = 8;
constexpr int kDH = 64;
constexpr int kL  = 6;
constexpr int kV  = 32000;
constexpr int kBS = kB * kS;          // 2048 rows
#define EPS_ 1e-5f

// ---------------- WMMA types ----------------
typedef __attribute__((ext_vector_type(16))) _Float16 v16h;
typedef __attribute__((ext_vector_type(8)))  float    v8f;

// ---------------- GEMM tiling ----------------
#define TM 128
#define TN 128
#define TK 32
// 256 threads = 8 wave32 waves, 2 (M) x 4 (N); each wave: 64x32 C tile
// = 4 M-frags x 2 N-frags, K-step 32 -> 8 v_wmma per step per wave.

// ---------------- TDM support detection ----------------
#if __has_builtin(__builtin_amdgcn_tensor_load_to_lds) && \
    __has_builtin(__builtin_amdgcn_s_wait_tensorcnt)
#define USE_TDM 1
#else
#define USE_TDM 0
#endif

#if USE_TDM
typedef unsigned int u32x4_t __attribute__((ext_vector_type(4)));
typedef int          i32x4_t __attribute__((ext_vector_type(4)));
typedef int          i32x8_t __attribute__((ext_vector_type(8)));

// One 2D TDM descriptor: tile = TK x TM (elements x rows), f16 (2B),
// tensor_dimN = remaining extent from the tile origin (TDM zero-fills OOB).
__device__ inline void tdm_load_2d(const _Float16* gptr, int ld_elems,
                                   long long d0_rem, long long d1_rem,
                                   const _Float16* lds_dst) {
  const unsigned long long ga = (unsigned long long)(const void*)gptr;
  // flat pointers to LDS carry the LDS byte offset in the low 32 bits
  const unsigned lo = (unsigned)(unsigned long long)(const void*)lds_dst;
  u32x4_t g0 = { 1u,                                   // count=1, user mode
                 lo,                                   // lds_addr
                 (unsigned)ga,                         // global_addr[31:0]
                 ((unsigned)((ga >> 32) & 0x01ffffffu)) | 0x80000000u }; // type=2
  const unsigned d0 = (unsigned)d0_rem;
  const unsigned d1 = (unsigned)d1_rem;
  i32x8_t g1 = {
    (int)0x00010000u,                                  // data_size=1 -> 2 bytes
    (int)((d0 & 0xffffu) << 16),                       // tensor_dim0 lo16
    (int)(((d0 >> 16) & 0xffffu) | ((d1 & 0xffffu) << 16)),
    (int)(((d1 >> 16) & 0xffffu) | ((unsigned)TK << 16)), // tile_dim0 = 32
    (int)TM,                                           // tile_dim1 = 128
    (int)ld_elems,                                     // tensor_dim0_stride
    0, 0
  };
  i32x4_t z4 = {0, 0, 0, 0};
#if __clang_major__ >= 23
  i32x8_t z8 = {0, 0, 0, 0, 0, 0, 0, 0};
  __builtin_amdgcn_tensor_load_to_lds(g0, g1, z4, z4, z8, 0);
#else
  __builtin_amdgcn_tensor_load_to_lds(g0, g1, z4, z4, 0);
#endif
}
#endif  // USE_TDM

__device__ inline v8f zero8() {
  v8f v;
#pragma unroll
  for (int i = 0; i < 8; ++i) v[i] = 0.f;
  return v;
}

// A fragment (16x32 f16). Lanes 0-15: row M=lane, K={0..7,16..23};
// lanes 16-31: row M=lane-16, K={8..15,24..31}.
__device__ inline v16h load_frag_a(const _Float16* sA, int m0, int lane) {
  const int row = m0 + (lane & 15);
  const int kb  = (lane >> 4) * 8;
  const _Float16* p = sA + row * TK;
  v16h a;
#pragma unroll
  for (int j = 0; j < 8; ++j) {
    a[j]     = p[kb + j];
    a[j + 8] = p[kb + 16 + j];
  }
  return a;
}

// B fragment (32x16 f16), sB stored [n][k]. Lanes 0-15: col=lane, K={0..7,16..23};
// lanes 16-31: col=lane-16, K={8..15,24..31}.
__device__ inline v16h load_frag_b(const _Float16* sB, int n0, int lane) {
  const int col = n0 + (lane & 15);
  const int kb  = (lane >> 4) * 8;
  const _Float16* p = sB + col * TK + kb;
  v16h b;
#pragma unroll
  for (int j = 0; j < 8; ++j) {
    b[j]     = p[j];
    b[j + 8] = p[16 + j];
  }
  return b;
}

__device__ inline float gelu_exact(float x) {
  return 0.5f * x * (1.0f + erff(x * 0.70710678118654752f));
}

// ============ f16 NT GEMM: C = act(alpha * A(f16,[M,K]) x B(f16,[N,K])^T + bias)
struct Gemm16P {
  const _Float16* A;
  const _Float16* Bm;
  const float*    bias;   // nullable
  float*          C;      // f32 out
  _Float16*       Ch;     // optional f16 mirror (nullable), same ldc/strides
  int M, N, K, lda, ldb, ldc;
  long long sAo, sAi, sBo, sBi, sCo, sCi;   // batch offsets (elements)
  int zdiv, act;          // act: 0 none, 1 exact GELU
  float alpha;
};

__global__ __launch_bounds__(256) void gemm16_kernel(Gemm16P p) {
  __shared__ _Float16 sA[2][TM * TK];
  __shared__ _Float16 sB[2][TN * TK];

  const int tid  = threadIdx.x;
  const int lane = tid & 31;
  const int w    = tid >> 5;
  const int wr   = w >> 2;     // 0..1
  const int wc   = w & 3;      // 0..3
  const int m0   = blockIdx.y * TM;
  const int n0   = blockIdx.x * TN;
  const int zo   = (int)blockIdx.z / p.zdiv;
  const int zi   = (int)blockIdx.z % p.zdiv;

  const _Float16* Ap = p.A  + (long long)zo * p.sAo + (long long)zi * p.sAi;
  const _Float16* Bp = p.Bm + (long long)zo * p.sBo + (long long)zi * p.sBi;
  float*          Cp = p.C  + (long long)zo * p.sCo + (long long)zi * p.sCi;
  _Float16*       Hp = p.Ch ? p.Ch + (long long)zo * p.sCo + (long long)zi * p.sCi
                            : (_Float16*)0;

  v8f acc[4][2];
#pragma unroll
  for (int i = 0; i < 4; ++i)
#pragma unroll
    for (int j = 0; j < 2; ++j) acc[i][j] = zero8();

  auto stage_tiles = [&](int buf, int kk) {
#if USE_TDM
    if (tid < 32) {   // wave 0 drives the DMA; EXEC is ignored by TDM
      tdm_load_2d(Ap + (long long)m0 * p.lda + kk, p.lda,
                  (long long)p.K - kk, (long long)p.M - m0, &sA[buf][0]);
      tdm_load_2d(Bp + (long long)n0 * p.ldb + kk, p.ldb,
                  (long long)p.K - kk, (long long)p.N - n0, &sB[buf][0]);
    }
#else
    // fallback: vectorized 16B copies; clamp B rows (cols >= N never stored)
#pragma unroll
    for (int i = 0; i < 2; ++i) {
      const int c = tid + i * 256;          // 512 chunks of 8 halfs
      const int r = c >> 2, kc = (c & 3) * 8;
      *(uint4*)&sA[buf][r * TK + kc] =
          *(const uint4*)&Ap[((long long)m0 + r) * p.lda + kk + kc];
    }
#pragma unroll
    for (int i = 0; i < 2; ++i) {
      const int c = tid + i * 256;
      const int r = c >> 2, kc = (c & 3) * 8;
      long long gn = (long long)n0 + r;
      if (gn >= p.N) gn = p.N - 1;          // safe dup; cols >= N are not written
      *(uint4*)&sB[buf][r * TK + kc] =
          *(const uint4*)&Bp[gn * p.ldb + kk + kc];
    }
#endif
  };

  auto stage_wait = [&](bool more) {
#if USE_TDM
    if (tid < 32) {
      if (more) __builtin_amdgcn_s_wait_tensorcnt(2);  // keep next tile in flight
      else      __builtin_amdgcn_s_wait_tensorcnt(0);
    }
#else
    (void)more;
#endif
  };

  const int nk = p.K / TK;        // K is always a multiple of TK here
  stage_tiles(0, 0);

  for (int kt = 0; kt < nk; ++kt) {
    const int pb = kt & 1;
    if (kt + 1 < nk) stage_tiles(1 - pb, (kt + 1) * TK);
    stage_wait(kt + 1 < nk);
    __syncthreads();

    const _Float16* cA = &sA[pb][0];
    const _Float16* cB = &sB[pb][0];
    v16h af[4];
#pragma unroll
    for (int mf = 0; mf < 4; ++mf)
      af[mf] = load_frag_a(cA, wr * 64 + mf * 16, lane);
    v16h bf[2];
#pragma unroll
    for (int nf = 0; nf < 2; ++nf)
      bf[nf] = load_frag_b(cB, wc * 32 + nf * 16, lane);

#pragma unroll
    for (int mf = 0; mf < 4; ++mf)
#pragma unroll
      for (int nf = 0; nf < 2; ++nf)
        acc[mf][nf] = __builtin_amdgcn_wmma_f32_16x16x32_f16(
            false, af[mf], false, bf[nf], (short)0, acc[mf][nf], false, false);

    __syncthreads();
  }

  // epilogue: VGPR r -> row (8*(lane>=16)+r), col (lane&15)
#pragma unroll
  for (int mf = 0; mf < 4; ++mf) {
#pragma unroll
    for (int nf = 0; nf < 2; ++nf) {
#pragma unroll
      for (int r = 0; r < 8; ++r) {
        const int row = m0 + wr * 64 + mf * 16 + ((lane >> 4) << 3) + r;
        const int col = n0 + wc * 32 + nf * 16 + (lane & 15);
        if (col < p.N) {                 // M is always a multiple of TM
          float v = acc[mf][nf][r] * p.alpha;
          if (p.bias) v += p.bias[col];
          if (p.act == 1) v = gelu_exact(v);
          Cp[(long long)row * p.ldc + col] = v;
          if (Hp) Hp[(long long)row * p.ldc + col] = (_Float16)v;
        }
      }
    }
  }
}

// ---------------- block reductions (256 threads, wave32) ----------------
__device__ inline float block_sum256(float v) {
  __shared__ float red[8];
  __syncthreads();
#pragma unroll
  for (int o = 16; o; o >>= 1) v += __shfl_xor(v, o, 32);
  const int wv = threadIdx.x >> 5, ln = threadIdx.x & 31;
  if (ln == 0) red[wv] = v;
  __syncthreads();
  float t = (ln < 8) ? red[ln] : 0.f;
#pragma unroll
  for (int o = 4; o; o >>= 1) t += __shfl_xor(t, o, 32);
  return __shfl(t, 0, 32);
}

__device__ inline float block_max256(float v) {
  __shared__ float red[8];
  __syncthreads();
#pragma unroll
  for (int o = 16; o; o >>= 1) v = fmaxf(v, __shfl_xor(v, o, 32));
  const int wv = threadIdx.x >> 5, ln = threadIdx.x & 31;
  if (ln == 0) red[wv] = v;
  __syncthreads();
  float t = (ln < 8) ? red[ln] : -3.0e38f;
#pragma unroll
  for (int o = 4; o; o >>= 1) t = fmaxf(t, __shfl_xor(t, o, 32));
  return __shfl(t, 0, 32);
}

// ---------------- conversions ----------------
__global__ __launch_bounds__(256) void convert_f16_kernel(const float* in,
                                                          _Float16* out,
                                                          long long n4) {
  const long long i = (long long)blockIdx.x * 256 + threadIdx.x;
  if (i >= n4) return;
  float4 v = ((const float4*)in)[i];
  _Float16* o = out + i * 4;
  o[0] = (_Float16)v.x; o[1] = (_Float16)v.y;
  o[2] = (_Float16)v.z; o[3] = (_Float16)v.w;
}

// out[c][r] = (f16) in[r][c] ; batched via blockIdx.z
__global__ __launch_bounds__(256) void transpose_convert_kernel(
    const float* in, _Float16* out, int R, int C, long long sIn, long long sOut) {
  __shared__ _Float16 tile[32][34];
  const float* ip = in + (long long)blockIdx.z * sIn;
  _Float16*    op = out + (long long)blockIdx.z * sOut;
  const int c0 = blockIdx.x * 32, r0 = blockIdx.y * 32;
  const int tx = threadIdx.x & 31, ty = threadIdx.x >> 5;
#pragma unroll
  for (int i = 0; i < 4; ++i) {
    const int r = ty + i * 8;
    tile[r][tx] = (_Float16)ip[(long long)(r0 + r) * C + c0 + tx];
  }
  __syncthreads();
#pragma unroll
  for (int i = 0; i < 4; ++i) {
    const int c = ty + i * 8;
    op[(long long)(c0 + c) * R + r0 + tx] = tile[tx][c];
  }
}

// ---------------- embedding: z/zh = w_emb[tok] + p_emb[pos] ----------------
__global__ __launch_bounds__(128) void embed_kernel(const int* tokens,
                                                    const float* w_emb,
                                                    const float* p_emb,
                                                    float* z, _Float16* zh) {
  const int bs  = blockIdx.x;
  const int pos = bs % kS;
  const int tok = tokens[bs];
  const float4 a = ((const float4*)(w_emb + (long long)tok * kH))[threadIdx.x];
  const float4 p = ((const float4*)(p_emb + (long long)pos * kH))[threadIdx.x];
  float4 o;
  o.x = a.x + p.x; o.y = a.y + p.y; o.z = a.z + p.z; o.w = a.w + p.w;
  ((float4*)(z + (long long)bs * kH))[threadIdx.x] = o;
  _Float16* zo = zh + (long long)bs * kH + threadIdx.x * 4;
  zo[0] = (_Float16)o.x; zo[1] = (_Float16)o.y;
  zo[2] = (_Float16)o.z; zo[3] = (_Float16)o.w;
}

// ---------------- masked softmax on scores [B,NH,S,S] (+ f16 mirror) ----------
__global__ __launch_bounds__(256) void attn_softmax_kernel(float* sc,
                                                           _Float16* sc16,
                                                           const int* lengths) {
  const int i = blockIdx.x, h = blockIdx.y, b = blockIdx.z;
  const int len = lengths[b];
  const long long off = ((((long long)b * kNH + h) * kS) + i) * kS;
  float* row = sc + off;
  const int j = threadIdx.x;
  const bool ok = (i < len) && (j <= i) && (j < len);
  const float v = ok ? row[j] : -1e30f;
  const float m = block_max256(v);
  const float e = ok ? expf(v - m) : 0.f;
  const float den = block_sum256(e);
  const float r = (den > 0.f) ? (e / den) : 0.f;
  row[j] = r;
  sc16[off + j] = (_Float16)r;
}

// ---------------- head-mean attention -> out [B,S,S] ----------------
__global__ __launch_bounds__(256) void att_mean_kernel(const float* sc, float* out) {
  const long long idx = (long long)blockIdx.x * 256 + threadIdx.x;
  const long long total = (long long)kB * kS * kS;
  if (idx >= total) return;
  const int j = (int)(idx % kS);
  const long long t = idx / kS;
  const int i = (int)(t % kS);
  const int b = (int)(t / kS);
  float s = 0.f;
#pragma unroll
  for (int h = 0; h < kNH; ++h)
    s += sc[(((long long)b * kNH + h) * kS + i) * kS + j];
  out[idx] = s * (1.0f / kNH);
}

// ---------------- residual + LayerNorm (+ f16 mirror) ----------------
__global__ __launch_bounds__(256) void add_ln_kernel(const float* x, const float* y,
                                                     const float* g, const float* be,
                                                     float* out, _Float16* out16) {
  const int row = blockIdx.x;
  const float* xr = x + (long long)row * kH;
  const float* yr = y + (long long)row * kH;
  const float v0 = xr[threadIdx.x]       + yr[threadIdx.x];
  const float v1 = xr[threadIdx.x + 256] + yr[threadIdx.x + 256];
  const float mu = block_sum256(v0 + v1) * (1.0f / kH);
  const float d0 = v0 - mu, d1 = v1 - mu;
  const float var = block_sum256(d0 * d0 + d1 * d1) * (1.0f / kH);
  const float rs = rsqrtf(var + EPS_);
  const float o0 = d0 * rs * g[threadIdx.x]       + be[threadIdx.x];
  const float o1 = d1 * rs * g[threadIdx.x + 256] + be[threadIdx.x + 256];
  float* orow = out + (long long)row * kH;
  orow[threadIdx.x]       = o0;
  orow[threadIdx.x + 256] = o1;
  if (out16) {
    _Float16* hrow = out16 + (long long)row * kH;
    hrow[threadIdx.x]       = (_Float16)o0;
    hrow[threadIdx.x + 256] = (_Float16)o1;
  }
}

// ---------------- vocab softmax over V=32000, in place ----------------
__global__ __launch_bounds__(256) void vocab_softmax_kernel(float* y) {
  const int row = blockIdx.x;
  float* r = y + (long long)row * kV;
  float m = -3.0e38f;
  for (int j = threadIdx.x; j < kV; j += 256) m = fmaxf(m, r[j]);
  m = block_max256(m);
  float s = 0.f;
  for (int j = threadIdx.x; j < kV; j += 256) s += expf(r[j] - m);
  s = block_sum256(s);
  const float inv = 1.0f / s;
  for (int j = threadIdx.x; j < kV; j += 256) r[j] = expf(r[j] - m) * inv;
}

// ---------------- host-side GEMM launcher ----------------
static inline void launch_gemm16(hipStream_t st, const _Float16* A, const _Float16* Bm,
                                 const float* bias, float* C, _Float16* Ch,
                                 int M, int N, int K, int lda, int ldb, int ldc,
                                 int act, float alpha, int batch = 1, int zdiv = 1,
                                 long long sAo = 0, long long sAi = 0,
                                 long long sBo = 0, long long sBi = 0,
                                 long long sCo = 0, long long sCi = 0) {
  Gemm16P p;
  p.A = A; p.Bm = Bm; p.bias = bias; p.C = C; p.Ch = Ch;
  p.M = M; p.N = N; p.K = K; p.lda = lda; p.ldb = ldb; p.ldc = ldc;
  p.sAo = sAo; p.sAi = sAi; p.sBo = sBo; p.sBi = sBi; p.sCo = sCo; p.sCi = sCi;
  p.zdiv = zdiv; p.act = act; p.alpha = alpha;
  dim3 grid((N + TN - 1) / TN, (M + TM - 1) / TM, batch);
  gemm16_kernel<<<grid, dim3(256), 0, st>>>(p);
}

extern "C" void kernel_launch(void* const* d_in, const int* in_sizes, int n_in,
                              void* d_out, int out_size, void* d_ws, size_t ws_size,
                              hipStream_t stream) {
  (void)in_sizes; (void)n_in; (void)out_size; (void)ws_size;

  const int*   tokens  = (const int*)  d_in[0];
  const int*   lengths = (const int*)  d_in[1];
  const float* w_emb   = (const float*)d_in[2];
  const float* p_emb   = (const float*)d_in[3];
  const float* Wk      = (const float*)d_in[4];
  const float* bk      = (const float*)d_in[5];
  const float* W1      = (const float*)d_in[6];
  const float* b1      = (const float*)d_in[7];
  const float* W2      = (const float*)d_in[8];
  const float* b2      = (const float*)d_in[9];
  const float* g1      = (const float*)d_in[10];
  const float* be1     = (const float*)d_in[11];
  const float* g2      = (const float*)d_in[12];
  const float* be2     = (const float*)d_in[13];
  const float* fc_w    = (const float*)d_in[14];
  const float* fc_b    = (const float*)d_in[15];

  float* y      = (float*)d_out;                      // [B,S,V]
  float* attOut = y + (long long)kB * kS * kV;        // [B,S,S]

  // ---- workspace carve (bump allocator, 256B aligned) ----
  char* wsp = (char*)d_ws;
  auto alloc = [&](size_t bytes) -> char* {
    char* r = wsp;
    wsp += (bytes + 255) & ~(size_t)255;
    return r;
  };
  const long long BSH = (long long)kBS * kH;          // 1 Mi elements
  const long long SC  = (long long)kB * kNH * kS * kS;
  float* z      = (float*)alloc(BSH * 4);
  float* kbuf   = (float*)alloc(BSH * 4);
  float* aout   = (float*)alloc(BSH * 4);
  float* ff2    = (float*)alloc(BSH * 4);
  float* scores = (float*)alloc(SC * 4);
  _Float16* zh     = (_Float16*)alloc(BSH * 2);
  _Float16* kh     = (_Float16*)alloc(BSH * 2);       // kproj / ff1 f16
  _Float16* lnh    = (_Float16*)alloc(BSH * 2);
  _Float16* khT    = (_Float16*)alloc(BSH * 2);       // [B, H, S]
  _Float16* att16  = (_Float16*)alloc(SC * 2);
  _Float16* WkT    = (_Float16*)alloc((long long)kL * kH * kH * 2);
  _Float16* W1T    = (_Float16*)alloc((long long)kL * kH * kH * 2);
  _Float16* W2T    = (_Float16*)alloc((long long)kL * kH * kH * 2);
  _Float16* fcwT   = (_Float16*)alloc((long long)kH * kH * 2);
  _Float16* wemb16 = (_Float16*)alloc((long long)kV * kH * 2);

  const float scale = 1.0f / sqrtf((float)kH);        // sqrt(H), per reference
  const long long HH = (long long)kH * kH;

  // ---- operand conversion (once per call) ----
  {
    const long long n4 = (long long)kV * kH / 4;
    convert_f16_kernel<<<dim3((unsigned)((n4 + 255) / 256)), dim3(256), 0, stream>>>(
        w_emb, wemb16, n4);
  }
  transpose_convert_kernel<<<dim3(kH / 32, kH / 32, kL), dim3(256), 0, stream>>>(
      Wk, WkT, kH, kH, HH, HH);
  transpose_convert_kernel<<<dim3(kH / 32, kH / 32, kL), dim3(256), 0, stream>>>(
      W1, W1T, kH, kH, HH, HH);
  transpose_convert_kernel<<<dim3(kH / 32, kH / 32, kL), dim3(256), 0, stream>>>(
      W2, W2T, kH, kH, HH, HH);
  transpose_convert_kernel<<<dim3(kH / 32, kH / 32, 1), dim3(256), 0, stream>>>(
      fc_w, fcwT, kH, kH, 0, 0);

  embed_kernel<<<dim3(kBS), dim3(128), 0, stream>>>(tokens, w_emb, p_emb, z, zh);

  for (int l = 0; l < kL; ++l) {
    // k = z @ Wk^T? (weights pre-transposed: NT GEMM) -> kbuf (f32) + kh (f16)
    launch_gemm16(stream, zh, WkT + (long long)l * HH, bk + (long long)l * kH,
                  kbuf, kh, kBS, kH, kH, kH, kH, kH, 0, 1.0f);

    // khT[b, :, :] = kh[b, :, :]^T  (for att@V B operand)
    transpose_convert_kernel<<<dim3(kH / 32, kS / 32, kB), dim3(256), 0, stream>>>(
        kbuf, khT, kS, kH, (long long)kS * kH, (long long)kH * kS);

    // scores = scale * K_h x K_h^T per (b,h): NT, [256,64]x[256,64]^T
    launch_gemm16(stream, kh, kh, nullptr, scores, nullptr,
                  kS, kS, kDH, kH, kH, kS, 0, scale,
                  kB * kNH, kNH,
                  (long long)kS * kH, (long long)kDH,
                  (long long)kS * kH, (long long)kDH,
                  (long long)kNH * kS * kS, (long long)kS * kS);

    attn_softmax_kernel<<<dim3(kS, kNH, kB), dim3(256), 0, stream>>>(
        scores, att16, lengths);

    if (l == kL - 1) {
      const int total = kB * kS * kS;
      att_mean_kernel<<<dim3((total + 255) / 256), dim3(256), 0, stream>>>(scores, attOut);
    }

    // attn_out = att @ K_h : NT with khT  [256,256] x [64,256]^T
    launch_gemm16(stream, att16, khT, nullptr, aout, nullptr,
                  kS, kDH, kS, kS, kS, kH, 0, 1.0f,
                  kB * kNH, kNH,
                  (long long)kNH * kS * kS, (long long)kS * kS,
                  (long long)kH * kS, (long long)kDH * kS,
                  (long long)kS * kH, (long long)kDH);

    // ln1 = LN(z + attn_out) -> aout (f32) + lnh (f16)
    add_ln_kernel<<<dim3(kBS), dim3(256), 0, stream>>>(
        z, aout, g1 + (long long)l * kH, be1 + (long long)l * kH, aout, lnh);

    // ff1 = gelu(ln1 @ W1 + b1) -> kh (f16 mirror is the consumed one)
    launch_gemm16(stream, lnh, W1T + (long long)l * HH, b1 + (long long)l * kH,
                  kbuf, kh, kBS, kH, kH, kH, kH, kH, 1, 1.0f);

    // ff2 = ff1 @ W2 + b2 -> ff2 (f32)
    launch_gemm16(stream, kh, W2T + (long long)l * HH, b2 + (long long)l * kH,
                  ff2, nullptr, kBS, kH, kH, kH, kH, kH, 0, 1.0f);

    // z = LN(ln1 + ff2) -> z (f32) + zh (f16)
    add_ln_kernel<<<dim3(kBS), dim3(256), 0, stream>>>(
        aout, ff2, g2 + (long long)l * kH, be2 + (long long)l * kH, z, zh);
  }

  // h = z @ fc_w + fc_b -> kbuf (f32) + kh (f16)
  launch_gemm16(stream, zh, fcwT, fc_b, kbuf, kh,
                kBS, kH, kH, kH, kH, kH, 0, 1.0f);

  // logits = h @ w_emb^T -> d_out   [2048,512] x [32000,512]^T (NT)
  launch_gemm16(stream, kh, wemb16, nullptr, y, nullptr,
                kBS, kV, kH, kH, kH, kV, 0, 1.0f);

  // y = softmax(logits) over V, in place
  vocab_softmax_kernel<<<dim3(kBS), dim3(256), 0, stream>>>(y);
}
